// SimpleGNNLayer_76605036691533
// MI455X (gfx1250) — matrix operand order repaired and checked by reading, not compiled, not run on hardware
//
#include <hip/hip_runtime.h>
#include <hip/hip_bf16.h>

typedef __bf16 v16bf __attribute__((ext_vector_type(16)));
typedef float  v8f   __attribute__((ext_vector_type(8)));

// ---------- helpers ----------
__device__ __forceinline__ unsigned short f2bf(float f) {
    unsigned u = __float_as_uint(f);
    u = (u + 0x7FFFu + ((u >> 16) & 1u)) >> 16;   // round-to-nearest-even
    return (unsigned short)u;
}
__device__ __forceinline__ float bf2f(unsigned short h) {
    return __uint_as_float(((unsigned)h) << 16);
}
// order-preserving float -> uint key for atomicMax-based segment max
__device__ __forceinline__ unsigned keyenc(float f) {
    unsigned u = __float_as_uint(f);
    return (u & 0x80000000u) ? ~u : (u | 0x80000000u);
}
__device__ __forceinline__ float keydec(unsigned k) {
    return __uint_as_float((k & 0x80000000u) ? (k ^ 0x80000000u) : ~k);
}

union Frag { uint4 q[2]; v16bf v; };

// A fragment (16x32 bf16, M=rows): lane l holds row m=l&15.
// half=l>>4: run0 K = kt*32 + 8*half + [0..7], run1 K = run0 + 16. (ISA 7.12.2)
__device__ __forceinline__ v16bf load_a_frag(const unsigned short* base, int stride,
                                             int lane, int kt) {
    const int half = lane >> 4, m = lane & 15;
    const unsigned short* p = base + m * stride + kt * 32 + half * 8;
    Frag f;
    f.q[0] = *(const uint4*)(p);
    f.q[1] = *(const uint4*)(p + 16);
    return f.v;
}
// B fragment from pre-packed weights: 32 contiguous bytes per lane.
__device__ __forceinline__ v16bf load_b_frag(const unsigned short* wp, int ksteps,
                                             int nt, int kt, int lane) {
    const unsigned short* p = wp + ((size_t)((nt * ksteps + kt) * 32 + lane) << 4);
    Frag f;
    f.q[0] = *(const uint4*)(p);
    f.q[1] = *(const uint4*)(p + 8);
    return f.v;
}
__device__ __forceinline__ v8f wmma_bf16(v16bf a, v16bf b, v8f c) {
    return __builtin_amdgcn_wmma_f32_16x16x32_bf16(false, a, false, b, (short)0, c,
                                                   false, false);
}
__device__ __forceinline__ void st_bf4(unsigned short* p, float4 v) {
    uint2 q;
    q.x = (unsigned)f2bf(v.x) | ((unsigned)f2bf(v.y) << 16);
    q.y = (unsigned)f2bf(v.z) | ((unsigned)f2bf(v.w) << 16);
    *(uint2*)p = q;
}

// ---------- weight pack: f32 [K,N] row-major -> bf16 WMMA-B tile layout ----------
// t = ((nt*Ks + kt)*32 + l)*16 + e ; n = nt*16 + (l&15) ; k = kt*32 + 8*(l>>4) + (e&7) + 16*(e>>3)
__global__ __launch_bounds__(256) void pack_w_kernel(const float* __restrict__ W,
                                                     unsigned short* __restrict__ Wp,
                                                     int K, int Nn) {
    int t = blockIdx.x * 256 + threadIdx.x;
    if (t >= K * Nn) return;
    int Ks = K >> 5;
    int e = t & 15;
    int l = (t >> 4) & 31;
    int tile = t >> 9;
    int kt = tile % Ks;
    int nt = tile / Ks;
    int n = nt * 16 + (l & 15);
    int k = kt * 32 + 8 * (l >> 4) + (e & 7) + 16 * (e >> 3);
    Wp[t] = f2bf(W[(size_t)k * Nn + n]);
}

// ---------- pass 1: per-edge MLP + attention logits ----------
// 1 wave handles 32 edges = two M=16 tiles sharing every B fragment
// => 2x weight-stream reuse + two independent WMMA accumulator chains.
#define SA 264   // (256+8) bf16 elems -> 528B row stride, 16B aligned
#define SH 136   // (128+8)
__global__ __launch_bounds__(32) void edge_kernel(
    const float* __restrict__ nf, const int* __restrict__ srcI,
    const int* __restrict__ dstI, const int* __restrict__ etype,
    const float* __restrict__ rel,
    const unsigned short* __restrict__ w1p, const float* __restrict__ b1,
    const unsigned short* __restrict__ w2p, const float* __restrict__ b2,
    const unsigned short* __restrict__ aw1p, const float* __restrict__ ab1,
    const float* __restrict__ aw2, const float* __restrict__ ab2,
    unsigned short* __restrict__ msg_out, float* __restrict__ attn_raw,
    unsigned* __restrict__ maxkey, int E) {
    __shared__ unsigned short A[32 * SA];   // src|rel, later dst|msg (bf16)
    __shared__ unsigned short H[32 * SH];   // hidden (bf16)
    __shared__ float attnAcc[32];

    const int l = threadIdx.x;
    const int half = l >> 4, ln = l & 15;
    const int e0 = blockIdx.x * 32;
    if (e0 >= E) return;
    const int valid = min(32, E - e0);

    unsigned short* A0 = A;
    unsigned short* A1 = A + 16 * SA;
    unsigned short* H0 = H;
    unsigned short* H1 = H + 16 * SH;

    // ---- stage A = [src_feat | rel_feat] as bf16 ----
    for (int m = 0; m < 32; ++m) {
        int e = e0 + ((m < valid) ? m : 0);
        int s = srcI[e];
        int ty = etype[e];
        float4 v = ((const float4*)(nf + (size_t)s * 128))[l];
        st_bf4(&A[m * SA + l * 4], v);
        float4 r = ((const float4*)(rel + (size_t)ty * 128))[l];
        st_bf4(&A[m * SA + 128 + l * 4], r);
    }
    __syncthreads();

    // ---- GEMM1: relu([32,256] @ msg_w1 + b1) -> H[32,128] ----
    for (int nt = 0; nt < 8; ++nt) {
        v8f acc0, acc1;
        float bias = b1[ln + 16 * nt];
#pragma unroll
        for (int i = 0; i < 8; ++i) { acc0[i] = bias; acc1[i] = bias; }
#pragma unroll
        for (int kt = 0; kt < 8; ++kt) {
            v16bf b = load_b_frag(w1p, 8, nt, kt, l);
            acc0 = wmma_bf16(load_a_frag(A0, SA, l, kt), b, acc0);
            acc1 = wmma_bf16(load_a_frag(A1, SA, l, kt), b, acc1);
        }
#pragma unroll
        for (int r = 0; r < 8; ++r) {
            float t0 = acc0[r], t1 = acc1[r];
            t0 = t0 > 0.f ? t0 : 0.f;
            t1 = t1 > 0.f ? t1 : 0.f;
            H0[(r + 8 * half) * SH + ln + 16 * nt] = f2bf(t0);
            H1[(r + 8 * half) * SH + ln + 16 * nt] = f2bf(t1);
        }
    }
    __syncthreads();

    // ---- restage A[:,0:128] = dst_feat (src no longer needed) ----
    for (int m = 0; m < 32; ++m) {
        int e = e0 + ((m < valid) ? m : 0);
        int d = dstI[e];
        float4 v = ((const float4*)(nf + (size_t)d * 128))[l];
        st_bf4(&A[m * SA + l * 4], v);
    }
    __syncthreads();

    // ---- GEMM2: messages = H @ msg_w2 + b2 ; spill bf16 + stage into A[:,128:256] ----
    for (int nt = 0; nt < 8; ++nt) {
        v8f acc0, acc1;
        float bias = b2[ln + 16 * nt];
#pragma unroll
        for (int i = 0; i < 8; ++i) { acc0[i] = bias; acc1[i] = bias; }
#pragma unroll
        for (int kt = 0; kt < 4; ++kt) {
            v16bf b = load_b_frag(w2p, 4, nt, kt, l);
            acc0 = wmma_bf16(load_a_frag(H0, SH, l, kt), b, acc0);
            acc1 = wmma_bf16(load_a_frag(H1, SH, l, kt), b, acc1);
        }
#pragma unroll
        for (int r = 0; r < 8; ++r) {
            int m = r + 8 * half;
            int n = ln + 16 * nt;
            unsigned short bf0 = f2bf(acc0[r]);
            unsigned short bf1 = f2bf(acc1[r]);
            A0[m * SA + 128 + n] = bf0;
            A1[m * SA + 128 + n] = bf1;
            if (m < valid)      msg_out[(size_t)(e0 + m) * 128 + n]      = bf0;
            if (m + 16 < valid) msg_out[(size_t)(e0 + 16 + m) * 128 + n] = bf1;
        }
    }
    __syncthreads();

    // ---- GEMM3: leaky_relu([dst|msg] @ attn_w1 + ab1, 0.2) . attn_w2 ----
    float s0[8], s1[8];
#pragma unroll
    for (int r = 0; r < 8; ++r) { s0[r] = 0.f; s1[r] = 0.f; }
    for (int nt = 0; nt < 8; ++nt) {
        v8f acc0, acc1;
        float bias = ab1[ln + 16 * nt];
#pragma unroll
        for (int i = 0; i < 8; ++i) { acc0[i] = bias; acc1[i] = bias; }
#pragma unroll
        for (int kt = 0; kt < 8; ++kt) {
            v16bf b = load_b_frag(aw1p, 8, nt, kt, l);
            acc0 = wmma_bf16(load_a_frag(A0, SA, l, kt), b, acc0);
            acc1 = wmma_bf16(load_a_frag(A1, SA, l, kt), b, acc1);
        }
        float w2v = aw2[ln + 16 * nt];
#pragma unroll
        for (int r = 0; r < 8; ++r) {
            float t0 = acc0[r], t1 = acc1[r];
            t0 = t0 > 0.f ? t0 : 0.2f * t0;
            t1 = t1 > 0.f ? t1 : 0.2f * t1;
            s0[r] += t0 * w2v;
            s1[r] += t1 * w2v;
        }
    }
    // reduce across the 16 lanes of each half (each half owns distinct M rows)
#pragma unroll
    for (int off = 1; off < 16; off <<= 1)
#pragma unroll
        for (int r = 0; r < 8; ++r) {
            s0[r] += __shfl_xor(s0[r], off, 32);
            s1[r] += __shfl_xor(s1[r], off, 32);
        }
    if (ln == 0) {
#pragma unroll
        for (int r = 0; r < 8; ++r) {
            attnAcc[half * 8 + r]      = s0[r] + ab2[0];
            attnAcc[16 + half * 8 + r] = s1[r] + ab2[0];
        }
    }
    __syncthreads();

    if (l < valid) {
        float raw = attnAcc[l];
        int e = e0 + l;
        attn_raw[e] = raw;
        atomicMax(&maxkey[dstI[e]], keyenc(raw));
    }
}

// ---------- pass 2: exp, segment-sum, unnormalized scatter-add (1 wave per edge) ----------
__global__ __launch_bounds__(256) void scatter_kernel(
    const unsigned short* __restrict__ msg, const float* __restrict__ attn_raw,
    const unsigned* __restrict__ maxkey, const int* __restrict__ dstI,
    float* __restrict__ sum, float* __restrict__ aggr, int E) {
    const int l = threadIdx.x & 31;
    const int e = blockIdx.x * 8 + (threadIdx.x >> 5);
    if (e >= E) return;
    const int d = dstI[e];
    const float ex = __expf(attn_raw[e] - keydec(maxkey[d]));
    if (l == 0) atomicAdd(&sum[d], ex);
    uint2 mm = *(const uint2*)(msg + (size_t)e * 128 + l * 4);
    float* ap = aggr + (size_t)d * 128 + l * 4;
    atomicAdd(ap + 0, bf2f((unsigned short)(mm.x & 0xFFFF)) * ex);
    atomicAdd(ap + 1, bf2f((unsigned short)(mm.x >> 16)) * ex);
    atomicAdd(ap + 2, bf2f((unsigned short)(mm.y & 0xFFFF)) * ex);
    atomicAdd(ap + 3, bf2f((unsigned short)(mm.y >> 16)) * ex);
}

// ---------- pass 3: LN1 -> FFN (WMMA, paired-nt dual chains) -> LN2 ----------
#define SHF 520  // (512+8)
__global__ __launch_bounds__(32) void node_kernel(
    const float* __restrict__ nf, const float* __restrict__ aggr,
    const float* __restrict__ sum,
    const float* __restrict__ g1, const float* __restrict__ be1,
    const unsigned short* __restrict__ f1p, const float* __restrict__ fb1,
    const unsigned short* __restrict__ f2p, const float* __restrict__ fb2,
    const float* __restrict__ g2, const float* __restrict__ be2,
    float* __restrict__ out, int N) {
    __shared__ unsigned short Xb[16 * SH];   // x as bf16 (A of FFN1)
    __shared__ float Xf[16 * 128];           // x f32 for residual
    __shared__ unsigned short Hf[16 * SHF];  // FFN hidden bf16
    __shared__ float F[16 * 128];            // FFN out f32

    const int l = threadIdx.x;
    const int half = l >> 4, ln = l & 15;
    const int n0 = blockIdx.x * 16;
    if (n0 >= N) return;
    const int valid = min(16, N - n0);

    // ---- x = LN1(node + aggr/(sum+eps)) ----
    for (int m = 0; m < 16; ++m) {
        int row = n0 + ((m < valid) ? m : 0);
        float inv = 1.f / (sum[row] + 1e-8f);
        float4 a = ((const float4*)(nf + (size_t)row * 128))[l];
        float4 g = ((const float4*)(aggr + (size_t)row * 128))[l];
        float f[4] = {a.x + g.x * inv, a.y + g.y * inv, a.z + g.z * inv, a.w + g.w * inv};
        float s1 = f[0] + f[1] + f[2] + f[3];
        float s2 = f[0]*f[0] + f[1]*f[1] + f[2]*f[2] + f[3]*f[3];
#pragma unroll
        for (int off = 16; off > 0; off >>= 1) {
            s1 += __shfl_xor(s1, off, 32);
            s2 += __shfl_xor(s2, off, 32);
        }
        float mean = s1 * (1.f / 128.f);
        float var = s2 * (1.f / 128.f) - mean * mean;
        float rstd = rsqrtf(var + 1e-5f);
        float4 xb;
        float* xb4 = &xb.x;
#pragma unroll
        for (int j = 0; j < 4; ++j) {
            int idx = l * 4 + j;
            float x = (f[j] - mean) * rstd * g1[idx] + be1[idx];
            Xf[m * 128 + idx] = x;
            xb4[j] = x;
        }
        st_bf4(&Xb[m * SH + l * 4], xb);
    }
    __syncthreads();

    // ---- FFN1: gelu([16,128] @ ffn_w1 + fb1) -> Hf[16,512] ----
    for (int nt = 0; nt < 32; nt += 2) {
        v8f acc0, acc1;
        float bias0 = fb1[ln + 16 * nt];
        float bias1 = fb1[ln + 16 * (nt + 1)];
#pragma unroll
        for (int i = 0; i < 8; ++i) { acc0[i] = bias0; acc1[i] = bias1; }
#pragma unroll
        for (int kt = 0; kt < 4; ++kt) {
            v16bf a = load_a_frag(Xb, SH, l, kt);
            acc0 = wmma_bf16(a, load_b_frag(f1p, 4, nt, kt, l), acc0);
            acc1 = wmma_bf16(a, load_b_frag(f1p, 4, nt + 1, kt, l), acc1);
        }
#pragma unroll
        for (int r = 0; r < 8; ++r) {
            float t0 = acc0[r], t1 = acc1[r];
            float ge0 = 0.5f * t0 * (1.0f + erff(t0 * 0.70710678118f));  // exact gelu
            float ge1 = 0.5f * t1 * (1.0f + erff(t1 * 0.70710678118f));
            Hf[(r + 8 * half) * SHF + ln + 16 * nt]       = f2bf(ge0);
            Hf[(r + 8 * half) * SHF + ln + 16 * (nt + 1)] = f2bf(ge1);
        }
    }
    __syncthreads();

    // ---- FFN2: Hf @ ffn_w2 + fb2 -> F[16,128] ----
    for (int nt = 0; nt < 8; nt += 2) {
        v8f acc0, acc1;
        float bias0 = fb2[ln + 16 * nt];
        float bias1 = fb2[ln + 16 * (nt + 1)];
#pragma unroll
        for (int i = 0; i < 8; ++i) { acc0[i] = bias0; acc1[i] = bias1; }
#pragma unroll
        for (int kt = 0; kt < 16; ++kt) {
            v16bf a = load_a_frag(Hf, SHF, l, kt);
            acc0 = wmma_bf16(a, load_b_frag(f2p, 16, nt, kt, l), acc0);
            acc1 = wmma_bf16(a, load_b_frag(f2p, 16, nt + 1, kt, l), acc1);
        }
#pragma unroll
        for (int r = 0; r < 8; ++r) {
            F[(r + 8 * half) * 128 + ln + 16 * nt]       = acc0[r];
            F[(r + 8 * half) * 128 + ln + 16 * (nt + 1)] = acc1[r];
        }
    }
    __syncthreads();

    // ---- out = LN2(x + ffn) ----
    for (int m = 0; m < valid; ++m) {
        float f[4];
#pragma unroll
        for (int j = 0; j < 4; ++j) {
            int idx = l * 4 + j;
            f[j] = F[m * 128 + idx] + Xf[m * 128 + idx];
        }
        float s1 = f[0] + f[1] + f[2] + f[3];
        float s2 = f[0]*f[0] + f[1]*f[1] + f[2]*f[2] + f[3]*f[3];
#pragma unroll
        for (int off = 16; off > 0; off >>= 1) {
            s1 += __shfl_xor(s1, off, 32);
            s2 += __shfl_xor(s2, off, 32);
        }
        float mean = s1 * (1.f / 128.f);
        float var = s2 * (1.f / 128.f) - mean * mean;
        float rstd = rsqrtf(var + 1e-5f);
#pragma unroll
        for (int j = 0; j < 4; ++j) {
            int idx = l * 4 + j;
            out[(size_t)(n0 + m) * 128 + idx] = (f[j] - mean) * rstd * g2[idx] + be2[idx];
        }
    }
}

// ---------- host ----------
extern "C" void kernel_launch(void* const* d_in, const int* in_sizes, int n_in,
                              void* d_out, int out_size, void* d_ws, size_t ws_size,
                              hipStream_t stream) {
    const float* nf   = (const float*)d_in[0];
    const int*   ei   = (const int*)d_in[1];
    const int*   ety  = (const int*)d_in[2];
    const float* rel  = (const float*)d_in[3];
    const float* mw1  = (const float*)d_in[4];
    const float* mb1  = (const float*)d_in[5];
    const float* mw2  = (const float*)d_in[6];
    const float* mb2  = (const float*)d_in[7];
    const float* aw1  = (const float*)d_in[8];
    const float* ab1  = (const float*)d_in[9];
    const float* aw2  = (const float*)d_in[10];
    const float* ab2  = (const float*)d_in[11];
    const float* l1g  = (const float*)d_in[12];
    const float* l1b  = (const float*)d_in[13];
    const float* fw1  = (const float*)d_in[14];
    const float* fb1  = (const float*)d_in[15];
    const float* fw2  = (const float*)d_in[16];
    const float* fb2  = (const float*)d_in[17];
    const float* l2g  = (const float*)d_in[18];
    const float* l2b  = (const float*)d_in[19];

    const int E = in_sizes[2];
    const int N = in_sizes[0] / 128;

    // workspace carve (256B aligned)
    char* p = (char*)d_ws;
    auto carve = [&](size_t bytes) -> void* {
        void* r = (void*)p;
        p += (bytes + 255) & ~(size_t)255;
        return r;
    };
    unsigned short* msg  = (unsigned short*)carve((size_t)E * 128 * 2);
    float*          araw = (float*)carve((size_t)E * 4);
    unsigned*       mkey = (unsigned*)carve((size_t)N * 4);
    float*          ssum = (float*)carve((size_t)N * 4);
    float*          aggr = (float*)carve((size_t)N * 128 * 4);
    unsigned short* w1p  = (unsigned short*)carve(256 * 128 * 2);
    unsigned short* w2p  = (unsigned short*)carve(128 * 128 * 2);
    unsigned short* aw1p = (unsigned short*)carve(256 * 128 * 2);
    unsigned short* f1p  = (unsigned short*)carve(128 * 512 * 2);
    unsigned short* f2p  = (unsigned short*)carve(512 * 128 * 2);

    hipMemsetAsync(mkey, 0, (size_t)N * 4, stream);
    hipMemsetAsync(ssum, 0, (size_t)N * 4, stream);
    hipMemsetAsync(aggr, 0, (size_t)N * 128 * 4, stream);

    pack_w_kernel<<<(256 * 128 + 255) / 256, 256, 0, stream>>>(mw1, w1p, 256, 128);
    pack_w_kernel<<<(128 * 128 + 255) / 256, 256, 0, stream>>>(mw2, w2p, 128, 128);
    pack_w_kernel<<<(256 * 128 + 255) / 256, 256, 0, stream>>>(aw1, aw1p, 256, 128);
    pack_w_kernel<<<(128 * 512 + 255) / 256, 256, 0, stream>>>(fw1, f1p, 128, 512);
    pack_w_kernel<<<(512 * 128 + 255) / 256, 256, 0, stream>>>(fw2, f2p, 512, 128);

    edge_kernel<<<(E + 31) / 32, 32, 0, stream>>>(
        nf, ei, ei + E, ety, rel, w1p, mb1, w2p, mb2, aw1p, ab1, aw2, ab2,
        msg, araw, mkey, E);

    scatter_kernel<<<(E + 7) / 8, 256, 0, stream>>>(msg, araw, mkey, ei + E, ssum, aggr, E);

    node_kernel<<<(N + 15) / 16, 32, 0, stream>>>(
        nf, aggr, ssum, l1g, l1b, f1p, fb1, f2p, fb2, l2g, l2b, (float*)d_out, N);
}
